// SparseMLP_21741124452874
// MI455X (gfx1250) — matrix operand order repaired and checked by reading, not compiled, not run on hardware
//
#include <hip/hip_runtime.h>
#include <math.h>

#define TOK 16384
#define HID 1024
#define FFNN 4096
#define NE 8

typedef __attribute__((ext_vector_type(16))) __bf16        v16bf;
typedef __attribute__((ext_vector_type(8)))  float         v8f;
typedef __attribute__((ext_vector_type(4)))  unsigned int  v4u;

union FragU { v4u u[2]; v16bf v; };

__device__ __forceinline__ unsigned short f2bf(float f) {
    unsigned int u = __float_as_uint(f);
    unsigned int r = u + 0x7FFFu + ((u >> 16) & 1u);   // round-to-nearest-even
    return (unsigned short)(r >> 16);
}

__device__ __forceinline__ float fast_tanh(float x) {
#if __has_builtin(__builtin_amdgcn_tanhf)
    return __builtin_amdgcn_tanhf(x);
#else
    float r;
    // gfx1250 hardware transcendental; v_nop covers the TRANS-op result hazard
    asm("v_tanh_f32 %0, %1\n\tv_nop" : "=v"(r) : "v"(x));
    return r;
#endif
}

__device__ __forceinline__ float gelu_tanh(float v) {
    const float c = 0.7978845608028654f;               // sqrt(2/pi)
    float t = c * (v + 0.044715f * v * v * v);
    return 0.5f * v * (1.0f + fast_tanh(t));
}

// ---------------- zero-fill the dummy A row used for boundary masking ---------
__global__ void zero_row(unsigned short* __restrict__ z) {
    int i = threadIdx.x * 4;                            // 256 threads x 4 = 1024
    z[i] = 0; z[i + 1] = 0; z[i + 2] = 0; z[i + 3] = 0;
}

// ---------------- fp32 -> bf16 elementwise convert (4 elems / thread) ---------
__global__ __launch_bounds__(256) void conv_f32_bf16(const float* __restrict__ in,
                                                     unsigned short* __restrict__ out,
                                                     int n) {
    int base = (blockIdx.x * 256 + threadIdx.x) * 4;
    if (base + 3 >= n) return;
    float4 v = *(const float4*)(in + base);
    out[base + 0] = f2bf(v.x);
    out[base + 1] = f2bf(v.y);
    out[base + 2] = f2bf(v.z);
    out[base + 3] = f2bf(v.w);
}

// ------------- w2 [E*F, H] fp32 -> per-expert transposed [E, H, F] bf16 -------
__global__ __launch_bounds__(256) void conv_transpose_w2(const float* __restrict__ in,
                                                         unsigned short* __restrict__ out) {
    __shared__ unsigned short tile[64][65];
    const int e  = blockIdx.z;
    const int f0 = blockIdx.x * 64;
    const int j0 = blockIdx.y * 64;
    #pragma unroll
    for (int i = 0; i < 16; ++i) {
        int lin = threadIdx.x + i * 256;
        int fl = lin >> 6, jl = lin & 63;
        float v = in[((size_t)(e * FFNN + f0 + fl)) * HID + (j0 + jl)];
        tile[fl][jl] = f2bf(v);
    }
    __syncthreads();
    #pragma unroll
    for (int i = 0; i < 16; ++i) {
        int lin = threadIdx.x + i * 256;
        int jl = lin >> 6, fl = lin & 63;
        out[((size_t)e * HID + (j0 + jl)) * FFNN + (f0 + fl)] = tile[fl][jl];
    }
}

// ------------- per-expert token ranges (ids sorted): offs[e] = lower_bound(e) -
__global__ void expert_offsets(const int* __restrict__ ids, int* __restrict__ offs,
                               int ntok, int ne) {
    int e = threadIdx.x;
    if (e > ne) return;
    int lo = 0, hi = ntok;
    while (lo < hi) {
        int mid = (lo + hi) >> 1;
        if (ids[mid] < e) lo = mid + 1; else hi = mid;
    }
    offs[e] = lo;
}

// ---------------- GEMM1: h = gelu(x @ w1_e^T), bf16 in / bf16 out -------------
// Block: 256 thr = 8 waves, tile 128(M) x 128(N). Wave = 32x64 (2x4 WMMA frags).
__global__ __launch_bounds__(256) void gemm1_gelu(const unsigned short* __restrict__ xb,
                                                  const unsigned short* __restrict__ w1b,
                                                  unsigned short* __restrict__ hb,
                                                  const int* __restrict__ offs,
                                                  const unsigned short* __restrict__ zrow) {
    const int e     = blockIdx.z;
    const int s     = offs[e];
    const int t_end = offs[e + 1];
    const int m0    = blockIdx.x * 128;
    if (m0 >= t_end || m0 + 128 <= s) return;           // tile does not touch expert e
    const int n0    = blockIdx.y * 128;

    const int lane = threadIdx.x & 31;
    const int w    = threadIdx.x >> 5;
    const int wm   = (w & 3) * 32;                      // 4 waves along M
    const int wn   = (w >> 2) * 64;                     // 2 waves along N
    const int half = lane >> 4;                         // 0 | 1
    const int l15  = lane & 15;
    const int kselA = half * 8;                         // A: K[0..8)+[16..24) | K[8..16)+[24..32)
    const int kselB = half * 8;                         // B: K[0..8)+[16..24) | K[8..16)+[24..32)

    // loop-invariant per-lane A base: invalid rows read the shared zero row
    const unsigned short* pA[2];
    #pragma unroll
    for (int mi = 0; mi < 2; ++mi) {
        int r = m0 + wm + mi * 16 + l15;
        bool valid = (r >= s && r < t_end);
        pA[mi] = valid ? (xb + (size_t)r * HID) : zrow;
    }
    const unsigned short* pB[4];
    #pragma unroll
    for (int ni = 0; ni < 4; ++ni) {
        int col = n0 + wn + ni * 16 + l15;
        pB[ni] = w1b + ((size_t)e * FFNN + col) * HID;
    }

    v8f acc[2][4] = {};

    for (int kk = 0; kk < HID; kk += 32) {
        v16bf a[2], b[4];
        #pragma unroll
        for (int mi = 0; mi < 2; ++mi) {
            FragU fa;
            const unsigned short* p = pA[mi] + kk + kselA;
            fa.u[0] = *(const v4u*)(p);
            fa.u[1] = *(const v4u*)(p + 16);
            a[mi] = fa.v;
        }
        #pragma unroll
        for (int ni = 0; ni < 4; ++ni) {
            FragU fb;
            const unsigned short* q = pB[ni] + kk + kselB;
            fb.u[0] = *(const v4u*)(q);
            fb.u[1] = *(const v4u*)(q + 16);
            b[ni] = fb.v;
        }
        #pragma unroll
        for (int mi = 0; mi < 2; ++mi)
            #pragma unroll
            for (int ni = 0; ni < 4; ++ni)
                acc[mi][ni] = __builtin_amdgcn_wmma_f32_16x16x32_bf16(
                    false, a[mi], false, b[ni], (short)0, acc[mi][ni], false, false);
    }

    // C layout: VGPR r -> lanes 0-15: M=r, N=lane ; lanes 16-31: M=r+8, N=lane-16
    #pragma unroll
    for (int mi = 0; mi < 2; ++mi)
        #pragma unroll
        for (int r = 0; r < 8; ++r) {
            int m = m0 + wm + mi * 16 + r + half * 8;
            if (m < s || m >= t_end) continue;
            #pragma unroll
            for (int ni = 0; ni < 4; ++ni) {
                int n = n0 + wn + ni * 16 + l15;
                hb[(size_t)m * FFNN + n] = f2bf(gelu_tanh(acc[mi][ni][r]));
            }
        }
}

// ---------------- GEMM2: out = h @ w2_e  (B pre-transposed to [E, H, F]) ------
__global__ __launch_bounds__(256) void gemm2_out(const unsigned short* __restrict__ hb,
                                                 const unsigned short* __restrict__ w2t,
                                                 float* __restrict__ out,
                                                 const int* __restrict__ offs) {
    const int e     = blockIdx.z;
    const int s     = offs[e];
    const int t_end = offs[e + 1];
    const int m0    = blockIdx.x * 128;
    if (m0 >= t_end || m0 + 128 <= s) return;
    const int n0    = blockIdx.y * 128;

    const int lane = threadIdx.x & 31;
    const int w    = threadIdx.x >> 5;
    const int wm   = (w & 3) * 32;
    const int wn   = (w >> 2) * 64;
    const int half = lane >> 4;
    const int l15  = lane & 15;
    const int kselA = half * 8;
    const int kselB = half * 8;

    const unsigned short* pA[2];
    #pragma unroll
    for (int mi = 0; mi < 2; ++mi)
        pA[mi] = hb + (size_t)(m0 + wm + mi * 16 + l15) * FFNN;   // rows always in-bounds
    const unsigned short* pB[4];
    #pragma unroll
    for (int ni = 0; ni < 4; ++ni)
        pB[ni] = w2t + ((size_t)e * HID + (n0 + wn + ni * 16 + l15)) * FFNN;

    v8f acc[2][4] = {};

    for (int kk = 0; kk < FFNN; kk += 32) {
        v16bf a[2], b[4];
        #pragma unroll
        for (int mi = 0; mi < 2; ++mi) {
            FragU fa;
            const unsigned short* p = pA[mi] + kk + kselA;
            fa.u[0] = *(const v4u*)(p);
            fa.u[1] = *(const v4u*)(p + 16);
            a[mi] = fa.v;                                // out-of-expert rows discarded at store
        }
        #pragma unroll
        for (int ni = 0; ni < 4; ++ni) {
            FragU fb;
            const unsigned short* q = pB[ni] + kk + kselB;
            fb.u[0] = *(const v4u*)(q);
            fb.u[1] = *(const v4u*)(q + 16);
            b[ni] = fb.v;
        }
        #pragma unroll
        for (int mi = 0; mi < 2; ++mi)
            #pragma unroll
            for (int ni = 0; ni < 4; ++ni)
                acc[mi][ni] = __builtin_amdgcn_wmma_f32_16x16x32_bf16(
                    false, a[mi], false, b[ni], (short)0, acc[mi][ni], false, false);
    }

    #pragma unroll
    for (int mi = 0; mi < 2; ++mi)
        #pragma unroll
        for (int r = 0; r < 8; ++r) {
            int m = m0 + wm + mi * 16 + r + half * 8;
            if (m < s || m >= t_end) continue;           // each token written by its own expert only
            #pragma unroll
            for (int ni = 0; ni < 4; ++ni) {
                int n = n0 + wn + ni * 16 + l15;
                out[(size_t)m * HID + n] = acc[mi][ni][r];
            }
        }
}

extern "C" void kernel_launch(void* const* d_in, const int* in_sizes, int n_in,
                              void* d_out, int out_size, void* d_ws, size_t ws_size,
                              hipStream_t stream) {
    const float* x    = (const float*)d_in[0];
    const float* w1   = (const float*)d_in[1];
    const float* w2   = (const float*)d_in[2];
    const int*   eids = (const int*)d_in[3];
    float*       out  = (float*)d_out;

    // workspace layout
    char* ws = (char*)d_ws;
    int*            offs = (int*)ws;                                   // 64 B
    unsigned short* zrow = (unsigned short*)(ws + 256);                // 2 KiB zero row
    unsigned short* xb   = (unsigned short*)(ws + 4096);               // 32 MiB
    unsigned short* w1b  = xb  + (size_t)TOK * HID;                    // 64 MiB
    unsigned short* w2t  = w1b + (size_t)NE * FFNN * HID;              // 64 MiB
    unsigned short* hb   = w2t + (size_t)NE * HID * FFNN;              // 128 MiB

    zero_row<<<1, 256, 0, stream>>>(zrow);
    conv_f32_bf16<<<(TOK * HID) / 1024, 256, 0, stream>>>(x, xb, TOK * HID);
    conv_f32_bf16<<<(NE * FFNN * HID) / 1024, 256, 0, stream>>>(w1, w1b, NE * FFNN * HID);
    conv_transpose_w2<<<dim3(FFNN / 64, HID / 64, NE), 256, 0, stream>>>(w2, w2t);
    expert_offsets<<<1, 32, 0, stream>>>(eids, offs, TOK, NE);
    gemm1_gelu<<<dim3(TOK / 128, FFNN / 128, NE), 256, 0, stream>>>(xb, w1b, hb, offs, zrow);
    gemm2_out<<<dim3(TOK / 128, HID / 128, NE), 256, 0, stream>>>(hb, w2t, out, offs);
}
